// GraphDiffusionModel_88467736363207
// MI455X (gfx1250) — matrix operand order repaired and checked by reading, not compiled, not run on hardware
//
#include <hip/hip_runtime.h>
#include <hip/hip_bf16.h>

// ---------------------------------------------------------------------------
// GraphDiffusionModel tower for MI455X (gfx1250), bf16 WMMA GEMM core.
// D = A*B+C via v_wmma_f32_16x16x32_bf16; f32 accumulate, bf16 data movement.
// Round 4: load both B fragments up front (distinct regs) so all fragment
//          ds_loads issue before the 4-WMMA chain.
// ---------------------------------------------------------------------------

typedef __attribute__((ext_vector_type(16))) __bf16 v16bf;
typedef __attribute__((ext_vector_type(8)))  float  v8f;
typedef unsigned short bf16b;   // raw bf16 bits
typedef unsigned int   u32b;

#if defined(__has_builtin)
#  if __has_builtin(__builtin_amdgcn_global_load_async_to_lds_b128) && \
      __has_builtin(__builtin_amdgcn_s_wait_asynccnt)
#    define USE_ASYNC_LDS 1
#  endif
#endif
#ifndef USE_ASYNC_LDS
#  define USE_ASYNC_LDS 0
#endif

#if USE_ASYNC_LDS
typedef int v4i32 __attribute__((vector_size(16)));
// 16B global->LDS async copy (tracked by ASYNCcnt, no VGPR round trip)
__device__ __forceinline__ void async_cp16(const void* g, void* l) {
  __builtin_amdgcn_global_load_async_to_lds_b128(
      (__attribute__((address_space(1))) v4i32*)
          (const __attribute__((address_space(1))) void*)g,
      (__attribute__((address_space(3))) v4i32*)
          (__attribute__((address_space(3))) void*)l,
      0, 0);
}
#endif

__device__ __forceinline__ bf16b f2bf(float f) {
  union { float f; u32b u; } v; v.f = f;
  u32b r = v.u + 0x7fffu + ((v.u >> 16) & 1u);   // round-to-nearest-even
  return (bf16b)(r >> 16);
}

union Frag16 { v16bf v; bf16b u[16]; uint4 q[2]; };

#define LDS_AS 40   // A tile row stride (bf16 elems): 80B rows, 16B aligned
#define LDS_BS 40   // B^T tile row stride

// ---- staging helpers -------------------------------------------------------
template <bool AF32>
__device__ __forceinline__ void gemm_issueA(const void* Ap, long lda, long rowBase,
                                            int ar, int ac, int k0,
                                            bf16b* dstA, float* fs, uint4* qa)
{
  if constexpr (AF32) {
    const float* ga = (const float*)Ap + (rowBase + ar) * lda + (k0 + ac);
    *(float4*)&fs[0]  = *(const float4*)(ga + 0);
    *(float4*)&fs[4]  = *(const float4*)(ga + 4);
    *(float4*)&fs[8]  = *(const float4*)(ga + 8);
    *(float4*)&fs[12] = *(const float4*)(ga + 12);
  } else {
    const bf16b* ga = (const bf16b*)Ap + (rowBase + ar) * lda + (k0 + ac);
#if USE_ASYNC_LDS
    bf16b* la = &dstA[ar * LDS_AS + ac];
    async_cp16(ga, la);
    async_cp16(ga + 8, la + 8);
#else
    qa[0] = *(const uint4*)ga;
    qa[1] = *(const uint4*)(ga + 8);
#endif
  }
}

template <bool AF32>
__device__ __forceinline__ void gemm_commitA(int ar, int ac, bf16b* dstA,
                                             const float* fs, const uint4* qa)
{
  bf16b* la = &dstA[ar * LDS_AS + ac];
  if constexpr (AF32) {
    union { uint4 q[2]; bf16b u[16]; } pk;
#pragma unroll
    for (int i = 0; i < 16; ++i) pk.u[i] = f2bf(fs[i]);
    *(uint4*)la       = pk.q[0];
    *(uint4*)(la + 8) = pk.q[1];
  } else {
#if !USE_ASYNC_LDS
    *(uint4*)la       = qa[0];
    *(uint4*)(la + 8) = qa[1];
#endif
  }
}

__device__ __forceinline__ void gemm_issueB(const bf16b* Bp, long ldb, long colBase,
                                            int bk, int bn, int k0, uint4* qb)
{
  *qb = *(const uint4*)(Bp + (long)(k0 + bk) * ldb + colBase + bn);
}

__device__ __forceinline__ void gemm_commitB(int bk, int bn, bf16b* dstB, uint4 qb)
{
  union { uint4 q; bf16b u[8]; } pb; pb.q = qb;
#pragma unroll
  for (int j = 0; j < 8; ++j) dstB[(bn + j) * LDS_BS + bk] = pb.u[j];
}

// ---------------------------------------------------------------------------
// Generic GEMM: C[M,Nc] = act(A[M,K] @ B[K,Nc] + bias)
//   AF32 : A operand is fp32 in global (converted to bf16 while staging)
//   OUTBF: store bf16 (else fp32)
//   RELU : apply max(0,.)
// Block = 256 threads (8 waves), tile 128(M) x 64(N), K step 32,
// double-buffered LDS, global prefetch overlapped with WMMA.
// ---------------------------------------------------------------------------
template <bool AF32, bool OUTBF, bool RELU>
__global__ __launch_bounds__(256) void k_gemm(
    const void* __restrict__ Ap, long lda,
    const bf16b* __restrict__ Bp, long ldb,
    void* __restrict__ Cp, long ldc,
    const float* __restrict__ bias, int K)
{
  __shared__ __align__(16) bf16b lAs[2][128 * LDS_AS];
  __shared__ __align__(16) bf16b lBs[2][64 * LDS_BS];   // transposed: [n][k]

  const int  tid  = threadIdx.x;
  const int  lane = tid & 31;
  const int  wave = tid >> 5;
  const int  hf   = lane >> 4;     // half-wave select
  const int  nl   = lane & 15;
  const long rowBase = (long)blockIdx.x * 128;
  const long colBase = (long)blockIdx.y * 64;
  const int  wm = (wave >> 1) * 32;
  const int  wn = (wave & 1) * 32;

  // staging assignments
  const int ar = tid >> 1;            // A: row 0..127
  const int ac = (tid & 1) * 16;      // A: 16 contiguous k
  const int bk = tid >> 3;            // B: k 0..31
  const int bn = (tid & 7) * 8;       // B: 8 contiguous n

  bf16b* curA = lAs[0]; bf16b* altA = lAs[1];
  bf16b* curB = lBs[0]; bf16b* altB = lBs[1];

  float fs[16]; uint4 qa[2]; uint4 qb;

  // prologue: stage K-tile 0 into buffer 0
  gemm_issueA<AF32>(Ap, lda, rowBase, ar, ac, 0, curA, fs, qa);
  gemm_issueB(Bp, ldb, colBase, bk, bn, 0, &qb);
  gemm_commitA<AF32>(ar, ac, curA, fs, qa);
  gemm_commitB(bk, bn, curB, qb);
#if USE_ASYNC_LDS
  if constexpr (!AF32) __builtin_amdgcn_s_wait_asynccnt(0);
#endif
  __syncthreads();

  v8f acc[2][2] = {};

  for (int k0 = 0; k0 < K; k0 += 32) {
    const bool hasNext = (k0 + 32) < K;
    // ---- start staging next K-tile into the alternate buffer ----
    if (hasNext) {
      gemm_issueA<AF32>(Ap, lda, rowBase, ar, ac, k0 + 32, altA, fs, qa);
      gemm_issueB(Bp, ldb, colBase, bk, bn, k0 + 32, &qb);
    }

    // ---- compute current tile: load all 4 fragments, then 4 WMMAs ----
    Frag16 af0, af1, bf0, bf1;
    {
      const bf16b* pa0 = &curA[(wm + nl) * LDS_AS + 8 * hf];
      af0.q[0] = *(const uint4*)pa0;
      af0.q[1] = *(const uint4*)(pa0 + 16);
      const bf16b* pa1 = &curA[(wm + 16 + nl) * LDS_AS + 8 * hf];
      af1.q[0] = *(const uint4*)pa1;
      af1.q[1] = *(const uint4*)(pa1 + 16);
      const bf16b* pb0 = &curB[(wn + nl) * LDS_BS + 8 * hf];
      bf0.q[0] = *(const uint4*)pb0;
      bf0.q[1] = *(const uint4*)(pb0 + 16);
      const bf16b* pb1 = &curB[(wn + 16 + nl) * LDS_BS + 8 * hf];
      bf1.q[0] = *(const uint4*)pb1;
      bf1.q[1] = *(const uint4*)(pb1 + 16);
    }
    acc[0][0] = __builtin_amdgcn_wmma_f32_16x16x32_bf16(
        false, af0.v, false, bf0.v, (short)0, acc[0][0], false, false);
    acc[1][0] = __builtin_amdgcn_wmma_f32_16x16x32_bf16(
        false, af1.v, false, bf0.v, (short)0, acc[1][0], false, false);
    acc[0][1] = __builtin_amdgcn_wmma_f32_16x16x32_bf16(
        false, af0.v, false, bf1.v, (short)0, acc[0][1], false, false);
    acc[1][1] = __builtin_amdgcn_wmma_f32_16x16x32_bf16(
        false, af1.v, false, bf1.v, (short)0, acc[1][1], false, false);

    // ---- commit staged regs into the alternate buffer ----
    if (hasNext) {
      gemm_commitA<AF32>(ar, ac, altA, fs, qa);
      gemm_commitB(bk, bn, altB, qb);
    }
#if USE_ASYNC_LDS
    if constexpr (!AF32) __builtin_amdgcn_s_wait_asynccnt(0);
#endif
    __syncthreads();
    bf16b* t;
    t = curA; curA = altA; altA = t;
    t = curB; curB = altB; altB = t;
  }

  // ---- epilogue: bias + relu + store (C layout: VGPR r -> M=r+8*hf, N=nl) ----
#pragma unroll
  for (int t = 0; t < 2; ++t) {
#pragma unroll
    for (int u = 0; u < 2; ++u) {
      const long gn = colBase + wn + 16 * u + nl;
      const float bv = bias ? bias[gn] : 0.0f;
#pragma unroll
      for (int r = 0; r < 8; ++r) {
        const long gm = rowBase + wm + 16 * t + hf * 8 + r;
        float v = acc[t][u][r] + bv;
        if (RELU) v = v > 0.0f ? v : 0.0f;
        if (OUTBF) ((bf16b*)Cp)[gm * ldc + gn] = f2bf(v);
        else       ((float*)Cp)[gm * ldc + gn] = v;
      }
    }
  }
}

// ---------------------------------------------------------------------------
// Tiny broadcast MLPs: h_t = mlp2(t1,t2,t) [128], h_cond = mlp2(c1,c2,c) [64]
// ---------------------------------------------------------------------------
__global__ void k_tiny(const float* __restrict__ t,
                       const float* __restrict__ Wt1, const float* __restrict__ bt1,
                       const float* __restrict__ Wt2, const float* __restrict__ bt2,
                       const float* __restrict__ c,
                       const float* __restrict__ Wc1, const float* __restrict__ bc1,
                       const float* __restrict__ Wc2, const float* __restrict__ bc2,
                       float* __restrict__ ht, float* __restrict__ hc)
{
  __shared__ float s1[128], s2[64];
  const int j = threadIdx.x;
  const float tv = t[0], cv = c[0];
  s1[j] = fmaxf(tv * Wt1[j] + bt1[j], 0.0f);
  if (j < 64) s2[j] = fmaxf(cv * Wc1[j] + bc1[j], 0.0f);
  __syncthreads();
  float a = bt2[j];
  for (int k = 0; k < 128; ++k) a += s1[k] * Wt2[k * 128 + j];
  ht[j] = fmaxf(a, 0.0f);
  if (j < 64) {
    float a2 = bc2[j];
    for (int k = 0; k < 64; ++k) a2 += s2[k] * Wc2[k * 64 + j];
    hc[j] = fmaxf(a2, 0.0f);
  }
}

// effective bias: out[j] = b[j] + sum v1[k]*W[(ofs1+k)*ldw+j] + sum v2[k]*W[(ofs2+k)*ldw+j]
__global__ void k_ebias(const float* __restrict__ W, long ldw,
                        const float* __restrict__ b,
                        const float* __restrict__ v1, long ofs1, int K1,
                        const float* __restrict__ v2, long ofs2, int K2,
                        float* __restrict__ out, int n)
{
  const int j = blockIdx.x * blockDim.x + threadIdx.x;
  if (j >= n) return;
  float a = b[j];
  for (int k = 0; k < K1; ++k) a += v1[k] * W[(ofs1 + k) * ldw + j];
  for (int k = 0; k < K2; ++k) a += v2[k] * W[(ofs2 + k) * ldw + j];
  out[j] = a;
}

__global__ void k_cvt(const float* __restrict__ in, bf16b* __restrict__ out, long n)
{
  long i = (long)blockIdx.x * blockDim.x + threadIdx.x;
  const long stride = (long)gridDim.x * blockDim.x;
  for (; i < n; i += stride) out[i] = f2bf(in[i]);
}

// h_Y0 = embY[Y] -> h_cat[:, 1536:1664) as bf16
__global__ void k_embed(const float* __restrict__ embY, const int* __restrict__ Y,
                        bf16b* __restrict__ hcat)
{
  const long row = blockIdx.x;
  const int  j = threadIdx.x;   // 128
  hcat[row * 2304 + 1536 + j] = f2bf(embY[(long)Y[row] * 128 + j]);
}

// HL[row, 0:384) = [hX_l | hY_l] gathered out of h_cat
__global__ void k_concat(const bf16b* __restrict__ hcat, int l, bf16b* __restrict__ hl)
{
  const long row = blockIdx.x;
  const int  j = threadIdx.x;   // 384
  const bf16b v = (j < 256) ? hcat[row * 2304 + 256 * l + j]
                            : hcat[row * 2304 + 1536 + 128 * l + (j - 256)];
  hl[row * 384 + j] = v;
}

// per-row layernorm of f32 [N,W] -> bf16 into h_cat slice
__global__ void k_layernorm(const float* __restrict__ X, int W,
                            const float* __restrict__ g, const float* __restrict__ b,
                            bf16b* __restrict__ out, long outStride, long colOfs)
{
  __shared__ float red[256];
  const int  j = threadIdx.x;
  const long row = blockIdx.x;
  const float x = X[row * W + j];
  red[j] = x;
  __syncthreads();
  for (int s = W >> 1; s > 0; s >>= 1) { if (j < s) red[j] += red[j + s]; __syncthreads(); }
  const float mean = red[0] * (1.0f / W);
  __syncthreads();
  const float d = x - mean;
  red[j] = d * d;
  __syncthreads();
  for (int s = W >> 1; s > 0; s >>= 1) { if (j < s) red[j] += red[j + s]; __syncthreads(); }
  const float var = red[0] * (1.0f / W);
  const float y = g[j] * d * rsqrtf(var + 1e-5f) + b[j];
  out[row * outStride + colOfs + j] = f2bf(y);
}

// ---------------------------------------------------------------------------
// Host orchestration
// ---------------------------------------------------------------------------
extern "C" void kernel_launch(void* const* d_in, const int* in_sizes, int n_in,
                              void* d_out, int out_size, void* d_ws, size_t ws_size,
                              hipStream_t stream)
{
  (void)in_sizes; (void)n_in; (void)out_size;

  // Flattened input order: top dict {t_float, X_t, Y, A_t, homophily_cond, params},
  // params pytree-flattened with sorted dict keys:
  //   c1{W,b}, c2{W,b}, embY, layers[0..4]{linX{W,b}, lnX{b,g}, linY{W,b}, lnY{b,g}},
  //   o1{W,b}, o2{W,b}, t1{W,b}, t2{W,b}, x1{W,b}, x2{W,b}
  const float* t_float = (const float*)d_in[0];
  const float* X_t     = (const float*)d_in[1];
  const int*   Y       = (const int*)  d_in[2];
  const float* A_t     = (const float*)d_in[3];
  const float* cond    = (const float*)d_in[4];
  const float* c1W = (const float*)d_in[5];
  const float* c1b = (const float*)d_in[6];
  const float* c2W = (const float*)d_in[7];
  const float* c2b = (const float*)d_in[8];
  const float* embY = (const float*)d_in[9];
  const float *lXW[5], *lXb[5], *lnXb[5], *lnXg[5], *lYW[5], *lYb[5], *lnYb[5], *lnYg[5];
  for (int l = 0; l < 5; ++l) {
    const int base = 10 + 8 * l;
    lXW[l]  = (const float*)d_in[base + 0];
    lXb[l]  = (const float*)d_in[base + 1];
    lnXb[l] = (const float*)d_in[base + 2];
    lnXg[l] = (const float*)d_in[base + 3];
    lYW[l]  = (const float*)d_in[base + 4];
    lYb[l]  = (const float*)d_in[base + 5];
    lnYb[l] = (const float*)d_in[base + 6];
    lnYg[l] = (const float*)d_in[base + 7];
  }
  const float* o1W = (const float*)d_in[50];
  const float* o1b = (const float*)d_in[51];
  const float* o2W = (const float*)d_in[52];
  const float* o2b = (const float*)d_in[53];
  const float* t1W = (const float*)d_in[54];
  const float* t1b = (const float*)d_in[55];
  const float* t2W = (const float*)d_in[56];
  const float* t2b = (const float*)d_in[57];
  const float* x1W = (const float*)d_in[58];
  const float* x1b = (const float*)d_in[59];
  const float* x2W = (const float*)d_in[60];
  const float* x2b = (const float*)d_in[61];

  // ---- workspace carve (256B aligned) ----
  uintptr_t cur = (uintptr_t)d_ws;
  auto alloc = [&](size_t bytes) -> void* {
    uintptr_t p = cur;
    cur = (cur + bytes + 255) & ~(uintptr_t)255;
    return (void*)p;
  };
  float* ht   = (float*)alloc(128 * 4);
  float* hc   = (float*)alloc(64 * 4);
  float* ebX  = (float*)alloc(5 * 256 * 4);
  float* ebY  = (float*)alloc(5 * 128 * 4);
  float* ebo1 = (float*)alloc(2496 * 4);
  float* T1   = (float*)alloc((size_t)8192 * 256 * 4);      // pre-LN scratch (X and Y reuse)
  bf16b* Wx1b = (bf16b*)alloc((size_t)512 * 256 * 2);
  bf16b* Wx2b = (bf16b*)alloc((size_t)256 * 256 * 2);
  bf16b* WXb  = (bf16b*)alloc((size_t)5 * 384 * 256 * 2);   // rows 0..383 of linX.W
  bf16b* WYb  = (bf16b*)alloc((size_t)5 * 128 * 128 * 2);   // rows 0..127 of linY.W
  bf16b* Wo1b = (bf16b*)alloc((size_t)2304 * 2496 * 2);     // rows 0..2303 of o1.W
  bf16b* Wo2b = (bf16b*)alloc((size_t)2496 * 512 * 2);
  bf16b* HCAT = (bf16b*)alloc((size_t)8192 * 2304 * 2);     // [hX0..hX5 | hY0..hY5] bf16
  bf16b* HL   = (bf16b*)alloc((size_t)8192 * 384 * 2);      // [hX_l | hY_l]
  bf16b* AGG  = (bf16b*)alloc((size_t)8192 * 384 * 2);      // A_t @ HL
  bf16b* H1   = (bf16b*)alloc((size_t)8192 * 256 * 2);      // x-MLP hidden
  bf16b* OH1  = (bf16b*)alloc((size_t)8192 * 2496 * 2);     // relu(o1(...))

  // One-time bf16 conversion of A_t if workspace allows (halves adjacency traffic
  // and lets the dominant GEMM take the async-to-LDS bf16 path).
  bf16b* A_tb = nullptr;
  {
    const size_t used = (size_t)(cur - (uintptr_t)d_ws);
    const size_t needA = (size_t)8192 * 8192 * 2;
    if (used + needA + 256 <= ws_size) A_tb = (bf16b*)alloc(needA);
  }

  const dim3 blk(256);

  // 1) broadcast MLPs
  k_tiny<<<1, 128, 0, stream>>>(t_float, t1W, t1b, t2W, t2b,
                                cond, c1W, c1b, c2W, c2b, ht, hc);

  // 2) bf16 weight conversion (+ optional one-time A_t conversion)
  k_cvt<<<256, 256, 0, stream>>>(x1W, Wx1b, (long)512 * 256);
  k_cvt<<<256, 256, 0, stream>>>(x2W, Wx2b, (long)256 * 256);
  for (int l = 0; l < 5; ++l) {
    k_cvt<<<256, 256, 0, stream>>>(lXW[l], WXb + (size_t)l * 384 * 256, (long)384 * 256);
    k_cvt<<<64, 256, 0, stream>>>(lYW[l], WYb + (size_t)l * 128 * 128, (long)128 * 128);
  }
  k_cvt<<<2048, 256, 0, stream>>>(o1W, Wo1b, (long)2304 * 2496);
  k_cvt<<<1024, 256, 0, stream>>>(o2W, Wo2b, (long)2496 * 512);
  if (A_tb) k_cvt<<<8192, 256, 0, stream>>>(A_t, A_tb, (long)8192 * 8192);

  // 3) effective biases (fold constant h_t / h_cond concat columns)
  for (int l = 0; l < 5; ++l) {
    k_ebias<<<1, 256, 0, stream>>>(lXW[l], 256, lXb[l], ht, 384, 128, hc, 512, 64,
                                   ebX + l * 256, 256);
    k_ebias<<<1, 128, 0, stream>>>(lYW[l], 128, lYb[l], ht, 0, 0, hc, 128, 64,
                                   ebY + l * 128, 128);
  }
  k_ebias<<<10, 256, 0, stream>>>(o1W, 2496, o1b, ht, 2304, 128, hc, 2432, 64, ebo1, 2496);

  // 4) feature MLP: h1 = relu(X_t@x1+b), hX0 = relu(h1@x2+b) -> HCAT[:,0:256)
  k_gemm<true,  true, true><<<dim3(64, 4), blk, 0, stream>>>(X_t, 512, Wx1b, 256, H1, 256, x1b, 512);
  k_gemm<false, true, true><<<dim3(64, 4), blk, 0, stream>>>(H1, 256, Wx2b, 256, HCAT, 2304, x2b, 256);

  // 5) hY0 = embY[Y] -> HCAT[:,1536:1664)
  k_embed<<<8192, 128, 0, stream>>>(embY, Y, HCAT);

  // 6) GNN layers
  for (int l = 0; l < 5; ++l) {
    k_concat<<<8192, 384, 0, stream>>>(HCAT, l, HL);
    // agg = A_t @ [hX_l | hY_l]   (M=8192, N=384, K=8192)
    if (A_tb)
      k_gemm<false, true, false><<<dim3(64, 6), blk, 0, stream>>>(A_tb, 8192, HL, 384, AGG, 384, nullptr, 8192);
    else
      k_gemm<true, true, false><<<dim3(64, 6), blk, 0, stream>>>(A_t, 8192, HL, 384, AGG, 384, nullptr, 8192);
    // hX_{l+1} = LN(relu(agg@Wx[0:384] + ebX))
    k_gemm<false, false, true><<<dim3(64, 4), blk, 0, stream>>>(AGG, 384, WXb + (size_t)l * 384 * 256, 256,
                                                               T1, 256, ebX + l * 256, 384);
    k_layernorm<<<8192, 256, 0, stream>>>(T1, 256, lnXg[l], lnXb[l], HCAT, 2304, 256 * (l + 1));
    // hY_{l+1} = LN(relu(agg[:,256:384]@Wy[0:128] + ebY))
    k_gemm<false, false, true><<<dim3(64, 2), blk, 0, stream>>>(AGG + 256, 384, WYb + (size_t)l * 128 * 128, 128,
                                                               T1, 128, ebY + l * 128, 128);
    k_layernorm<<<8192, 128, 0, stream>>>(T1, 128, lnYg[l], lnYb[l], HCAT, 2304, 1536 + 128 * (l + 1));
  }

  // 7) head: OH1 = relu(h_cat@o1 + ebo1); out = OH1@o2 + o2.b
  k_gemm<false, true, true><<<dim3(64, 39), blk, 0, stream>>>(HCAT, 2304, Wo1b, 2496, OH1, 2496, ebo1, 2304);
  k_gemm<false, false, false><<<dim3(64, 8), blk, 0, stream>>>(OH1, 2496, Wo2b, 512, d_out, 512, o2b, 2496);
}
